// CSWinBlock_13391708029362
// MI455X (gfx1250) — matrix operand order repaired and checked by reading, not compiled, not run on hardware
//
#include <hip/hip_runtime.h>

// ---------------------------------------------------------------------------
// CSWin block for MI455X (gfx1250, wave32, WMMA f32_16x16x32_f16)
// ---------------------------------------------------------------------------

typedef _Float16 half_t;
typedef __attribute__((ext_vector_type(16))) _Float16 v16h;
typedef __attribute__((ext_vector_type(8)))  float    v8f;

#define BATCH 8
#define HW    128
#define LTOK  (HW*HW)          // 16384 tokens per image
#define BL    (BATCH*LTOK)     // 131072 rows
#define CDIM  128
#define DFF   512

// ---------------- WMMA helpers (CDNA5 ISA 7.12.2 layouts) -------------------

__device__ __forceinline__ v8f wmma_f16(v16h a, v16h b, v8f c) {
  // D = A(16x32 f16) * B(32x16 f16) + C(16x16 f32)
  return __builtin_amdgcn_wmma_f32_16x16x32_f16(false, a, false, b,
                                                (short)0, c, false, false);
}

// A-matrix 16x32 f16 fragment from LDS [rows][ld] (row-major).
// lanes 0-15: M=lane, K=0..15 ; lanes 16-31: same M rows,
// K=16..31 (contiguous order, matches B helpers).
__device__ __forceinline__ v16h load_frag_a(const half_t* p, int ld,
                                            int row0, int k0, int lane) {
  int m  = row0 + (lane & 15);
  int kb = (lane & 16) ? 16 : 0;
  const half_t* rp = p + m * ld + k0;
  v16h a;
#pragma unroll
  for (int i = 0; i < 16; ++i) a[i] = rp[kb + i];
  return a;
}

// B-matrix 32x16 f16 fragment where memory holds B transposed: p[n][k]
// (i.e. row n of p is column n of B).  lanes 0-15 carry K=0..15,
// lanes 16-31 carry K=16..31 (mirrors the sparse-B table).
__device__ __forceinline__ v16h load_frag_b_t(const half_t* p, int ld,
                                              int n0, int k0, int lane) {
  int n  = n0 + (lane & 15);
  int kb = (lane & 16) ? 16 : 0;
  const half_t* rp = p + n * ld + k0 + kb;
  v16h b;
#pragma unroll
  for (int i = 0; i < 16; ++i) b[i] = rp[i];
  return b;
}

// B fragment where memory is K-major: p[k][n] (e.g. V tile [token][d]).
__device__ __forceinline__ v16h load_frag_b_kmaj(const half_t* p, int ld,
                                                 int k0, int n0, int lane) {
  int n  = n0 + (lane & 15);
  int kb = (lane & 16) ? 16 : 0;
  v16h b;
#pragma unroll
  for (int i = 0; i < 16; ++i) b[i] = p[(k0 + kb + i) * ld + n];
  return b;
}

// ---------------- tiny utility kernels --------------------------------------

__global__ __launch_bounds__(256) void cvt_f32_f16(const float* __restrict__ s,
                                                   half_t* __restrict__ d, int n) {
  int i = blockIdx.x * 256 + threadIdx.x;
  if (i < n) d[i] = (half_t)s[i];
}

// LayerNorm over 128-wide rows, one wave per row, output f16.
__global__ __launch_bounds__(256)
void ln_f16_kernel(const float* __restrict__ x, const float* __restrict__ g,
                   const float* __restrict__ bb, half_t* __restrict__ y) {
  int wave = threadIdx.x >> 5, lane = threadIdx.x & 31;
  size_t row = (size_t)blockIdx.x * 8 + wave;
  const float4* xr = (const float4*)(x + row * CDIM);
  float4 v = xr[lane];
  float s = v.x + v.y + v.z + v.w;
#pragma unroll
  for (int m = 1; m < 32; m <<= 1) s += __shfl_xor(s, m, 32);
  float mean = s * (1.0f / 128.0f);
  float d0 = v.x - mean, d1 = v.y - mean, d2 = v.z - mean, d3 = v.w - mean;
  float vs = d0*d0 + d1*d1 + d2*d2 + d3*d3;
#pragma unroll
  for (int m = 1; m < 32; m <<= 1) vs += __shfl_xor(vs, m, 32);
  float rstd = rsqrtf(vs * (1.0f / 128.0f) + 1e-5f);
  float4 gv = ((const float4*)g)[lane];
  float4 bv = ((const float4*)bb)[lane];
  half_t* yr = y + row * CDIM + lane * 4;
  yr[0] = (half_t)(d0 * rstd * gv.x + bv.x);
  yr[1] = (half_t)(d1 * rstd * gv.y + bv.y);
  yr[2] = (half_t)(d2 * rstd * gv.z + bv.z);
  yr[3] = (half_t)(d3 * rstd * gv.w + bv.w);
}

// ---------------- generic WMMA GEMM -----------------------------------------
// out[n][m] = sum_k A[n][k] * Bw[m][k]   (A f16 [N][K], Bw f16 [M][K])
// EPI 0: store f16 ; EPI 1: +bias +res -> f32 ; EPI 2: +bias, GELU -> f16
#define LDT 72

template <int EPI>
__global__ __launch_bounds__(256)
void gemm_kernel(const half_t* __restrict__ A, const half_t* __restrict__ Bw,
                 const float* __restrict__ bias, const float* __restrict__ res,
                 void* __restrict__ outp, int K, int Mtot) {
  __shared__ half_t lA[128][LDT];
  __shared__ half_t lB[128][LDT];
  int tid = threadIdx.x;
  int lane = tid & 31, wave = tid >> 5;
  int wr = wave >> 1, wc = wave & 1;
  size_t n0 = (size_t)blockIdx.x * 128;
  int m0 = blockIdx.y * 128;
  v8f acc[2][4] = {};

  for (int k0 = 0; k0 < K; k0 += 64) {
#pragma unroll
    for (int i = 0; i < 4; ++i) {
      int idx = i * 256 + tid;
      int row = idx >> 3, c8 = (idx & 7) * 8;
      *(uint4*)&lA[row][c8] = *(const uint4*)(A + (n0 + row) * K + k0 + c8);
      *(uint4*)&lB[row][c8] = *(const uint4*)(Bw + (size_t)(m0 + row) * K + k0 + c8);
      if (k0 + 64 < K) {
        __builtin_prefetch(A + (n0 + row) * K + k0 + 64 + c8, 0, 3);
        __builtin_prefetch(Bw + (size_t)(m0 + row) * K + k0 + 64 + c8, 0, 3);
      }
    }
    __syncthreads();
#pragma unroll
    for (int kc = 0; kc < 2; ++kc) {
      v16h a0 = load_frag_a(&lA[0][0], LDT, wr * 32,      kc * 32, lane);
      v16h a1 = load_frag_a(&lA[0][0], LDT, wr * 32 + 16, kc * 32, lane);
#pragma unroll
      for (int ct = 0; ct < 4; ++ct) {
        v16h b = load_frag_b_t(&lB[0][0], LDT, wc * 64 + ct * 16, kc * 32, lane);
        acc[0][ct] = wmma_f16(a0, b, acc[0][ct]);
        acc[1][ct] = wmma_f16(a1, b, acc[1][ct]);
      }
    }
    __syncthreads();
  }

  int hi = (lane & 16) ? 8 : 0;
  int ml = lane & 15;
#pragma unroll
  for (int rt = 0; rt < 2; ++rt)
#pragma unroll
    for (int ct = 0; ct < 4; ++ct)
#pragma unroll
      for (int r = 0; r < 8; ++r) {
        size_t n = n0 + wr * 32 + rt * 16 + r + hi;
        int m = m0 + wc * 64 + ct * 16 + ml;
        float v = acc[rt][ct][r];
        if constexpr (EPI == 0) {
          ((half_t*)outp)[n * Mtot + m] = (half_t)v;
        } else if constexpr (EPI == 1) {
          v += bias[m] + res[n * Mtot + m];
          ((float*)outp)[n * Mtot + m] = v;
        } else {
          v += bias[m];
          v = 0.5f * v * (1.0f + erff(v * 0.70710678118654752f));
          ((half_t*)outp)[n * Mtot + m] = (half_t)v;
        }
      }
}

// ---------------- CSWin windowed attention + LePE ----------------------------

__device__ __forceinline__ float lepe_val(const half_t* Vh /*[256][32]*/,
                                          const float* cwl, int br, int t, int d) {
  int r, c, Hs, Ws;
  if (br == 0) { r = t >> 1; c = t & 1;   Hs = 128; Ws = 2; }
  else         { r = t >> 7; c = t & 127; Hs = 2;   Ws = 128; }
  float acc = 0.f;
#pragma unroll
  for (int dy = 0; dy < 3; ++dy) {
    int rr = r + dy - 1;
#pragma unroll
    for (int dx = 0; dx < 3; ++dx) {
      int cc = c + dx - 1;
      if (rr >= 0 && rr < Hs && cc >= 0 && cc < Ws)
        acc += cwl[d * 9 + dy * 3 + dx] * (float)Vh[(rr * Ws + cc) * 32 + d];
    }
  }
  return acc;
}

// One block per (batch, window, head, branch): 8*64*2*2 = 2048 blocks.
__global__ __launch_bounds__(128)
void attn_kernel(const half_t* __restrict__ qkv, // [BL][384] f16 (q|k|v)
                 const float* __restrict__ cw0, const float* __restrict__ cb0,
                 const float* __restrict__ cw1, const float* __restrict__ cb1,
                 half_t* __restrict__ att) {     // [BL][128] f16
  __shared__ half_t Qh[256][32];
  __shared__ half_t Kh[256][32];
  __shared__ half_t Vh[256][32];
  __shared__ half_t Pch[4][16][32];
  __shared__ float  cwl[288];
  __shared__ float  cbl[32];

  int bid = blockIdx.x;
  int br = bid & 1, h = (bid >> 1) & 1, w = (bid >> 2) & 63, b = bid >> 8;
  int tid = threadIdx.x, lane = tid & 31, wv = tid >> 5;
  size_t bL = (size_t)b * LTOK;
  int ch0 = br * 64 + h * 32;

  // Stage the window's Q/K/V tiles (256 tokens x 32 dims each).
#pragma unroll
  for (int tt = 0; tt < 2; ++tt) {
    int t = tid * 2 + tt;
    int L = (br == 0) ? ((t >> 1) * 128 + 2 * w + (t & 1)) : (w * 256 + t);
    size_t base = (bL + L) * 384 + ch0;
    const uint4* sq = (const uint4*)(qkv + base);
    const uint4* sk = (const uint4*)(qkv + base + 128);
    const uint4* sv = (const uint4*)(qkv + base + 256);
    uint4* dq = (uint4*)&Qh[t][0];
    uint4* dk = (uint4*)&Kh[t][0];
    uint4* dv = (uint4*)&Vh[t][0];
#pragma unroll
    for (int i = 0; i < 4; ++i) { dq[i] = sq[i]; dk[i] = sk[i]; dv[i] = sv[i]; }
  }
  {
    const float* cw = br ? cw1 : cw0;
    const float* cb = br ? cb1 : cb0;
    for (int i = tid; i < 288; i += 128) cwl[i] = cw[h * 288 + i];
    if (tid < 32) cbl[tid] = cb[h * 32 + tid];
  }
  __syncthreads();

  const float scale = 0.17677669529663687f; // 32^-0.5
  int hi = (lane & 16) ? 8 : 0;
  int nl = lane & 15;

  for (int blk = 0; blk < 4; ++blk) {
    int m0 = wv * 64 + blk * 16;
    v16h aq = load_frag_a(&Qh[0][0], 32, m0, 0, lane);
    v8f o0 = {}, o1 = {};
    float mrow[8], rsum[8];
#pragma unroll
    for (int r = 0; r < 8; ++r) { mrow[r] = -1e30f; rsum[r] = 0.f; }

    for (int jc = 0; jc < 8; ++jc) {           // 32-key chunks (flash style)
      v16h bk0 = load_frag_b_t(&Kh[0][0], 32, jc * 32,      0, lane);
      v16h bk1 = load_frag_b_t(&Kh[0][0], 32, jc * 32 + 16, 0, lane);
      v8f z = {};
      v8f s0 = wmma_f16(aq, bk0, z);
      v8f s1 = wmma_f16(aq, bk1, z);
#pragma unroll
      for (int r = 0; r < 8; ++r) {
        float a0 = s0[r] * scale, a1 = s1[r] * scale;
        float cm = fmaxf(a0, a1);
        cm = fmaxf(cm, __shfl_xor(cm, 1, 32));
        cm = fmaxf(cm, __shfl_xor(cm, 2, 32));
        cm = fmaxf(cm, __shfl_xor(cm, 4, 32));
        cm = fmaxf(cm, __shfl_xor(cm, 8, 32));
        float mn = fmaxf(mrow[r], cm);
        float corr = __expf(mrow[r] - mn);
        float p0 = __expf(a0 - mn), p1 = __expf(a1 - mn);
        float ps = p0 + p1;
        ps += __shfl_xor(ps, 1, 32);
        ps += __shfl_xor(ps, 2, 32);
        ps += __shfl_xor(ps, 4, 32);
        ps += __shfl_xor(ps, 8, 32);
        rsum[r] = rsum[r] * corr + ps;
        mrow[r] = mn;
        o0[r] *= corr; o1[r] *= corr;
        Pch[wv][r + hi][nl]      = (half_t)p0;   // re-layout C->A via LDS
        Pch[wv][r + hi][16 + nl] = (half_t)p1;
      }
#if __has_builtin(__builtin_amdgcn_sched_barrier)
      __builtin_amdgcn_sched_barrier(0);
#endif
      v16h ap  = load_frag_a(&Pch[wv][0][0], 32, 0, 0, lane);
      v16h bv0 = load_frag_b_kmaj(&Vh[0][0], 32, jc * 32, 0,  lane);
      v16h bv1 = load_frag_b_kmaj(&Vh[0][0], 32, jc * 32, 16, lane);
      o0 = wmma_f16(ap, bv0, o0);
      o1 = wmma_f16(ap, bv1, o1);
    }

    // Epilogue: normalize, add LePE (depthwise 3x3 on V) + bias, scatter out.
#pragma unroll
    for (int r = 0; r < 8; ++r) {
      int t = m0 + r + hi;
      int L = (br == 0) ? ((t >> 1) * 128 + 2 * w + (t & 1)) : (w * 256 + t);
      size_t obase = (bL + L) * CDIM + ch0;
      float inv = 1.0f / rsum[r];
      int d0 = nl, d1 = 16 + nl;
      float v0 = o0[r] * inv + cbl[d0] + lepe_val(&Vh[0][0], cwl, br, t, d0);
      float v1 = o1[r] * inv + cbl[d1] + lepe_val(&Vh[0][0], cwl, br, t, d1);
      att[obase + d0] = (half_t)v0;
      att[obase + d1] = (half_t)v1;
    }
  }
}

// ---------------- host-side launch ------------------------------------------

extern "C" void kernel_launch(void* const* d_in, const int* in_sizes, int n_in,
                              void* d_out, int out_size, void* d_ws, size_t ws_size,
                              hipStream_t stream) {
  (void)in_sizes; (void)n_in; (void)out_size; (void)ws_size;
  const float* x      = (const float*)d_in[0];
  const float* ln1_g  = (const float*)d_in[1];
  const float* ln1_b  = (const float*)d_in[2];
  const float* w_qkv  = (const float*)d_in[3];
  const float* w_proj = (const float*)d_in[4];
  const float* b_proj = (const float*)d_in[5];
  const float* cw0    = (const float*)d_in[6];
  const float* cb0    = (const float*)d_in[7];
  const float* cw1    = (const float*)d_in[8];
  const float* cb1    = (const float*)d_in[9];
  const float* ln2_g  = (const float*)d_in[10];
  const float* ln2_b  = (const float*)d_in[11];
  const float* w1     = (const float*)d_in[12];
  const float* b1     = (const float*)d_in[13];
  const float* w2     = (const float*)d_in[14];
  const float* b2     = (const float*)d_in[15];
  float* out = (float*)d_out;

  char* ws = (char*)d_ws;
  size_t off = 0;
  auto carve = [&](size_t bytes) -> void* {
    void* p = ws + off;
    off = (off + bytes + 255) & ~(size_t)255;
    return p;
  };
  half_t* wq16  = (half_t*)carve((size_t)384 * 128 * 2);
  half_t* wp16  = (half_t*)carve((size_t)128 * 128 * 2);
  half_t* w116  = (half_t*)carve((size_t)512 * 128 * 2);
  half_t* w216  = (half_t*)carve((size_t)128 * 512 * 2);
  half_t* y16   = (half_t*)carve((size_t)BL * 128 * 2);
  half_t* qkv16 = (half_t*)carve((size_t)BL * 384 * 2);
  half_t* att16 = (half_t*)carve((size_t)BL * 128 * 2);
  float*  xnew  = (float*) carve((size_t)BL * 128 * 4);
  half_t* h16   = (half_t*)carve((size_t)BL * 128 * 2);
  half_t* mid16 = (half_t*)carve((size_t)BL * 512 * 2);

  // 1) weights -> f16
  cvt_f32_f16<<<(384 * 128 + 255) / 256, 256, 0, stream>>>(w_qkv, wq16, 384 * 128);
  cvt_f32_f16<<<(128 * 128 + 255) / 256, 256, 0, stream>>>(w_proj, wp16, 128 * 128);
  cvt_f32_f16<<<(512 * 128 + 255) / 256, 256, 0, stream>>>(w1, w116, 512 * 128);
  cvt_f32_f16<<<(128 * 512 + 255) / 256, 256, 0, stream>>>(w2, w216, 128 * 512);

  // 2) LN1 -> f16
  ln_f16_kernel<<<BL / 8, 256, 0, stream>>>(x, ln1_g, ln1_b, y16);

  // 3) QKV GEMM: [BL,128] x [384,128]^T -> qkv16 [BL,384]
  gemm_kernel<0><<<dim3(BL / 128, 3), 256, 0, stream>>>(
      y16, wq16, nullptr, nullptr, qkv16, 128, 384);

  // 4) cross-shaped-window attention + LePE -> att16 [BL,128]
  attn_kernel<<<2048, 128, 0, stream>>>(qkv16, cw0, cb0, cw1, cb1, att16);

  // 5) proj + bias + residual(x) -> xnew f32
  gemm_kernel<1><<<dim3(BL / 128, 1), 256, 0, stream>>>(
      att16, wp16, b_proj, x, xnew, 128, 128);

  // 6) LN2 -> f16
  ln_f16_kernel<<<BL / 8, 256, 0, stream>>>(xnew, ln2_g, ln2_b, h16);

  // 7) MLP up + bias + exact GELU -> mid16 [BL,512]
  gemm_kernel<2><<<dim3(BL / 128, 4), 256, 0, stream>>>(
      h16, w116, b1, nullptr, mid16, 128, 512);

  // 8) MLP down + bias + residual(xnew) -> d_out f32
  gemm_kernel<1><<<dim3(BL / 128, 1), 256, 0, stream>>>(
      mid16, w216, b2, xnew, out, 512, 128);
}
